// DualPathGRU_13168369729787
// MI455X (gfx1250) — matrix operand order repaired and checked by reading, not compiled, not run on hardware
//
#include <hip/hip_runtime.h>
#include <hip/hip_bf16.h>

// ---------------------------------------------------------------------------
// DualPathGRU on MI455X (gfx1250): persistent wave32 WMMA kernel.
//   B=1024, S=512, H=512, IN=OUT=2, 3H=1536.
// 64 workgroups x 16 batch rows for the whole 512-step scan (8 wave32s each).
// h state lives in LDS (bf16, double buffered); W_hh/W1/W2 are pre-packed
// into v_wmma_f32_16x16x32_bf16 B-fragment layout in d_ws (gate-interleaved
// for W_hh) and streamed from L2 with an explicit 2-buffer ping-pong so B
// loads of step kb+1 overlap the WMMAs of step kb. Accumulators start from
// the inline-0 C operand; all biases are folded into the scalar epilogues.
// ---------------------------------------------------------------------------

typedef __attribute__((ext_vector_type(16))) __bf16 bf16x16;
typedef __attribute__((ext_vector_type(8)))  __bf16 bf16x8;
typedef __attribute__((ext_vector_type(8)))  float  f32x8;

#define WMMA_BF16(a, b, c) \
  __builtin_amdgcn_wmma_f32_16x16x32_bf16(false, (a), false, (b), (short)0, (c), false, false)

namespace dpg {
constexpr int S    = 512;
constexpr int H    = 512;
constexpr int OUTD = 2;
constexpr int H3   = 1536;
constexpr int BT   = 16;        // batch rows per workgroup (1 WMMA row block)
constexpr int ROWP = H + 8;     // padded LDS row stride for h (bf16 elems)
constexpr int ROWP2= 256 + 8;   // padded LDS row stride for y1 (bf16 elems)

// LDS carve (bytes, all 16B aligned)
constexpr int OFF_H0  = 0;
constexpr int SZ_H    = BT * ROWP * 2;              // 16640
constexpr int OFF_H1  = OFF_H0 + SZ_H;              // 16640
constexpr int OFF_Y1  = OFF_H1 + SZ_H;              // 33280
constexpr int SZ_Y1   = BT * ROWP2 * 2;             // 8448
constexpr int OFF_WIH = OFF_Y1 + SZ_Y1;             // 41728  (1536*4 f32)
constexpr int OFF_BIH = OFF_WIH + H3 * 4 * 4;       // 66304
constexpr int OFF_BHH = OFF_BIH + H3 * 4;           // 72448
constexpr int OFF_B1  = OFF_BHH + H3 * 4;           // 78592 (256 f32)
constexpr int OFF_B2  = OFF_B1 + 256 * 4;           // 79616 (pad to 16)
constexpr int OFF_XP  = OFF_B2 + 16;                // 79632 (16*2 f32)
constexpr int OFF_VS  = OFF_XP + BT * 2 * 4;        // 79760
constexpr int OFF_GIN = OFF_VS + BT * 2 * 4;        // 79888 (16*4 f32)
constexpr int SMEM_BYTES = OFF_GIN + BT * 4 * 4;    // 80144
} // namespace dpg

__device__ __forceinline__ bf16x16 join16(bf16x8 lo, bf16x8 hi) {
  return __builtin_shufflevector(lo, hi, 0,1,2,3,4,5,6,7,8,9,10,11,12,13,14,15);
}

// A fragment (16x32 bf16, contiguous K order k = 16h+s): 16 contiguous elems
// per lane; pA already includes lane-row and 8h offset, add another 8h.
__device__ __forceinline__ bf16x16 loadA(const __bf16* pA) {
  const __bf16* p = pA + ((__builtin_amdgcn_workitem_id_x() & 16) ? 8 : 0);
  return join16(*(const bf16x8*)(p), *(const bf16x8*)(p + 8));
}

__device__ __forceinline__ float fast_sigmoid(float x) {
  return __builtin_amdgcn_rcpf(1.0f + __expf(-x));
}
__device__ __forceinline__ float fast_tanh(float x) {
  const float e2 = __expf(2.0f * x);
  return 1.0f - 2.0f * __builtin_amdgcn_rcpf(e2 + 1.0f);
}

// ---------------------------------------------------------------------------
// B-fragment packing (contiguous K order k = 16h+e, same as loadA):
//   element e of lane l -> K = kb*32 + (l<16 ? e : 16+e), N = nb*16 + (l&15)
// ---------------------------------------------------------------------------

// W_hh [1536,512] -> gate-interleaved fragments: f = (jb*16+kb)*3 + gate,
// covering N-block nb = gate*32 + jb. R/Z/N fragments are contiguous in L2.
__global__ void prep_whh_frag(const float* __restrict__ src, __bf16* __restrict__ dst) {
  int tid = blockIdx.x * 256 + threadIdx.x;     // 1536*512 elems
  if (tid >= 1536 * 512) return;
  int e    = tid & 15;
  int lane = (tid >> 4) & 31;
  int f    = tid >> 9;          // 0..1535
  int gate = f % 3;
  int q    = f / 3;
  int kb   = q & 15;
  int jb   = q >> 4;            // 0..31
  int k    = kb * 32 + ((lane & 16) ? 16 + e : e);
  int n    = gate * 512 + jb * 16 + (lane & 15);
  dst[tid] = (__bf16)src[n * 512 + k];
}

// Row-major [N,512] -> plain fragments: f = nb*16 + kb (used for W1).
__global__ void prep_bfrag_k512(const float* __restrict__ src,
                                __bf16* __restrict__ dst, int total) {
  int tid = blockIdx.x * 256 + threadIdx.x;
  if (tid >= total) return;
  int e    = tid & 15;
  int lane = (tid >> 4) & 31;
  int f    = tid >> 9;
  int kb   = f & 15;
  int nb   = f >> 4;
  int k    = kb * 32 + ((lane & 16) ? 16 + e : e);
  int n    = nb * 16 + (lane & 15);
  dst[tid] = (__bf16)src[n * 512 + k];
}

// W2 [2,256]: pack B[k,n] = (n<2 ? W2[n,k] : 0) as 8 fragments (K=256,N=16pad).
__global__ void prep_w2frag(const float* __restrict__ W2, __bf16* __restrict__ dst) {
  int tid = blockIdx.x * 256 + threadIdx.x;   // 8 frags * 512 = 4096
  if (tid >= 8 * 512) return;
  int e    = tid & 15;
  int lane = (tid >> 4) & 31;
  int kb   = tid >> 9;
  int k    = kb * 32 + ((lane & 16) ? 16 + e : e);
  int n    = lane & 15;
  dst[tid] = (n < dpg::OUTD) ? (__bf16)W2[n * 256 + k] : (__bf16)0.0f;
}

// ---------------------------------------------------------------------------
// Persistent GRU scan kernel: 64 blocks x 256 threads (8 wave32s).
// ---------------------------------------------------------------------------
__global__ __launch_bounds__(256, 1) void gru_persist(
    const float* __restrict__ X0,  const float* __restrict__ V,
    const float* __restrict__ WihG,const float* __restrict__ bihG,
    const float* __restrict__ bhhG,const float* __restrict__ b1G,
    const float* __restrict__ b2G,
    const __bf16* __restrict__ fWhh, const __bf16* __restrict__ fW1,
    const __bf16* __restrict__ fW2,  float* __restrict__ out) {
  using namespace dpg;
  extern __shared__ __align__(16) char smem[];
  __bf16* hbuf0 = (__bf16*)(smem + OFF_H0);
  __bf16* hbuf1 = (__bf16*)(smem + OFF_H1);
  __bf16* y1    = (__bf16*)(smem + OFF_Y1);
  float*  wih   = (float*)(smem + OFF_WIH);
  float*  bih   = (float*)(smem + OFF_BIH);
  float*  bhh   = (float*)(smem + OFF_BHH);
  float*  b1    = (float*)(smem + OFF_B1);
  float*  b2    = (float*)(smem + OFF_B2);
  float*  xprev = (float*)(smem + OFF_XP);
  float*  vst   = (float*)(smem + OFF_VS);
  float*  gin   = (float*)(smem + OFF_GIN);

  const int tid   = threadIdx.x;
  const int lane  = tid & 31;
  const int w     = tid >> 5;            // wave id 0..7
  const int lrow  = lane & 15;
  const int off   = (lane & 16) ? 8 : 0; // A/C half-wave row offset
  const int batchBase = blockIdx.x * BT;

  // ---- one-time init: zero h0, cache small weights in LDS, load x_prev ----
  for (int i = tid; i < BT * ROWP; i += 256) hbuf0[i] = (__bf16)0.0f;
  for (int i = tid; i < H3 * 4;   i += 256) wih[i] = WihG[i];   // W_ih row-major [1536][4]
  for (int i = tid; i < H3;       i += 256) { bih[i] = bihG[i]; bhh[i] = bhhG[i]; }
  for (int i = tid; i < 256;      i += 256) b1[i] = b1G[i];
  if (tid < OUTD) b2[tid] = b2G[tid];
  if (tid < BT * OUTD) xprev[tid] = X0[batchBase * OUTD + tid];
  __syncthreads();

  for (int t = 0; t < S; ++t) {
    __bf16* hc = (t & 1) ? hbuf1 : hbuf0;   // h_{t-1}
    __bf16* hn = (t & 1) ? hbuf0 : hbuf1;   // h_t

    // ---- phase 0: stage v_t and gru_in = [v0 v1 x0 x1] ----
    if (tid < BT) {
      const float2 v = *(const float2*)(V + ((size_t)(batchBase + tid) * S + t) * 2);
      float* g = gin + tid * 4;
      g[0] = v.x; g[1] = v.y;
      g[2] = xprev[tid * 2 + 0]; g[3] = xprev[tid * 2 + 1];
      vst[tid * 2 + 0] = v.x; vst[tid * 2 + 1] = v.y;
    }
    __syncthreads();

    // ---- phase 1: gh = h @ W_hh^T (WMMA, ping-ponged A/B streams);
    //      fuse gi (K=4 input GEMM) + biases + gates + h update ----
#pragma unroll 2
    for (int jj = 0; jj < 4; ++jj) {
      const int jb  = w * 4 + jj;            // hidden-column block 0..31
      const int col = jb * 16 + lrow;        // hidden column j
      f32x8 aR = {}, aZ = {}, aN = {};       // inline-0 C operand
      const __bf16* pB = fWhh + (size_t)jb * 48 * 512 + (size_t)lane * 16;
      const __bf16* pA = hc + lrow * ROWP + off;
      bf16x16 A[2], BR[2], BZ[2], BN[2];
      A[0]  = loadA(pA);
      BR[0] = *(const bf16x16*)(pB);
      BZ[0] = *(const bf16x16*)(pB + 512);
      BN[0] = *(const bf16x16*)(pB + 1024);
#pragma unroll
      for (int kb = 0; kb < 16; ++kb) {
        const int cur = kb & 1, nxt = cur ^ 1;
        if (kb < 15) {                       // prefetch next K-step fragments
          A[nxt] = loadA(pA + (kb + 1) * 32);
          const __bf16* pb = pB + (size_t)(kb + 1) * 3 * 512;
          BR[nxt] = *(const bf16x16*)(pb);
          BZ[nxt] = *(const bf16x16*)(pb + 512);
          BN[nxt] = *(const bf16x16*)(pb + 1024);
        }
        aR = WMMA_BF16(A[cur], BR[cur], aR);
        aZ = WMMA_BF16(A[cur], BZ[cur], aZ);
        aN = WMMA_BF16(A[cur], BN[cur], aN);
      }
      // gi (K=4) + all biases + torch GRU cell update, per C element
      const float4 wr = *(const float4*)(wih + (size_t)(0   + col) * 4);
      const float4 wz = *(const float4*)(wih + (size_t)(512 + col) * 4);
      const float4 wn = *(const float4*)(wih + (size_t)(1024+ col) * 4);
      const float br = bih[col] + bhh[col];          // i_r + h_r biases
      const float bz = bih[512 + col] + bhh[512 + col];
      const float bn = bih[1024 + col];              // i_n bias only
      const float bhn = bhh[1024 + col];             // h_n bias (inside r*(.))
#pragma unroll
      for (int i = 0; i < 8; ++i) {
        const int row = off + i;
        const float4 g = *(const float4*)(gin + row * 4);
        const float ir = br + g.x * wr.x + g.y * wr.y + g.z * wr.z + g.w * wr.w;
        const float iz = bz + g.x * wz.x + g.y * wz.y + g.z * wz.z + g.w * wz.w;
        const float in_= bn + g.x * wn.x + g.y * wn.y + g.z * wn.z + g.w * wn.w;
        const float rg = fast_sigmoid(ir + aR[i]);
        const float zg = fast_sigmoid(iz + aZ[i]);
        const float ng = fast_tanh(in_ + rg * (aN[i] + bhn));
        const float hp = (float)hc[row * ROWP + col];
        hn[row * ROWP + col] = (__bf16)((1.0f - zg) * ng + zg * hp);
      }
    }
    __syncthreads();

    // ---- phase 2: y1 = relu(h_t @ W1^T + b1)  [16,512]x[512,256] ----
#pragma unroll
    for (int ii = 0; ii < 2; ++ii) {
      const int nb  = w * 2 + ii;            // 16 n-blocks over 8 waves
      const int col = nb * 16 + lrow;
      f32x8 acc = {};
      const __bf16* pB = fW1 + ((size_t)(nb * 16)) * 512 + (size_t)lane * 16;
      const __bf16* pA = hn + lrow * ROWP + off;
      bf16x16 A[2], Bb[2];
      A[0]  = loadA(pA);
      Bb[0] = *(const bf16x16*)(pB);
#pragma unroll
      for (int kb = 0; kb < 16; ++kb) {
        const int cur = kb & 1, nxt = cur ^ 1;
        if (kb < 15) {
          A[nxt]  = loadA(pA + (kb + 1) * 32);
          Bb[nxt] = *(const bf16x16*)(pB + (size_t)(kb + 1) * 512);
        }
        acc = WMMA_BF16(A[cur], Bb[cur], acc);
      }
      const float b1c = b1[col];
#pragma unroll
      for (int i = 0; i < 8; ++i) {
        const int row = off + i;
        y1[row * ROWP2 + col] = (__bf16)fmaxf(acc[i] + b1c, 0.0f);
      }
    }
    __syncthreads();

    // ---- phase 3: residual = y1 @ W2^T + b2 (zero-padded to 16 cols),
    //      x_t = x_prev + v_t + residual ; wave 0 only (uniform branch) ----
    if (w == 0) {
      f32x8 acc = {};
      const __bf16* pA = y1 + lrow * ROWP2 + off;
      const __bf16* pB = fW2 + (size_t)lane * 16;
      bf16x16 A[2], Bb[2];
      A[0]  = loadA(pA);
      Bb[0] = *(const bf16x16*)(pB);
#pragma unroll
      for (int kb = 0; kb < 8; ++kb) {
        const int cur = kb & 1, nxt = cur ^ 1;
        if (kb < 7) {
          A[nxt]  = loadA(pA + (kb + 1) * 32);
          Bb[nxt] = *(const bf16x16*)(pB + (size_t)(kb + 1) * 512);
        }
        acc = WMMA_BF16(A[cur], Bb[cur], acc);
      }
      if (lrow < OUTD) {                        // lanes 0,1,16,17 write
        const float b2c = b2[lrow];
#pragma unroll
        for (int i = 0; i < 8; ++i) {
          const int row = off + i;
          const float xt = xprev[row * 2 + lrow] + vst[row * 2 + lrow] + acc[i] + b2c;
          out[((size_t)(batchBase + row) * S + t) * OUTD + lrow] = xt;
          xprev[row * 2 + lrow] = xt;
        }
      }
    }
    __syncthreads();
  }
}

// ---------------------------------------------------------------------------
extern "C" void kernel_launch(void* const* d_in, const int* in_sizes, int n_in,
                              void* d_out, int out_size, void* d_ws, size_t ws_size,
                              hipStream_t stream) {
  (void)in_sizes; (void)n_in; (void)out_size; (void)ws_size;
  using namespace dpg;
  const float* X0  = (const float*)d_in[0];
  const float* V   = (const float*)d_in[1];
  const float* Wih = (const float*)d_in[2];
  const float* Whh = (const float*)d_in[3];
  const float* bih = (const float*)d_in[4];
  const float* bhh = (const float*)d_in[5];
  const float* W1  = (const float*)d_in[6];
  const float* b1  = (const float*)d_in[7];
  const float* W2  = (const float*)d_in[8];
  const float* b2  = (const float*)d_in[9];
  float* out = (float*)d_out;

  // workspace: bf16 WMMA B-fragments (rebuilt every call -> deterministic)
  __bf16* fWhh = (__bf16*)d_ws;              // 1536*512 (gate-interleaved)
  __bf16* fW1  = fWhh + (size_t)H3 * 512;    //  256*512
  __bf16* fW2  = fW1  + (size_t)256 * 512;   //    8*512

  prep_whh_frag<<<(H3 * 512 + 255) / 256, 256, 0, stream>>>(Whh, fWhh);
  prep_bfrag_k512<<<(256 * 512 + 255) / 256, 256, 0, stream>>>(W1, fW1, 256 * 512);
  prep_w2frag<<<(8 * 512 + 255) / 256, 256, 0, stream>>>(W2, fW2);

  (void)hipFuncSetAttribute((const void*)gru_persist,
                            hipFuncAttributeMaxDynamicSharedMemorySize,
                            SMEM_BYTES);
  gru_persist<<<1024 / BT, 256, SMEM_BYTES, stream>>>(
      X0, V, Wih, bih, bhh, b1, b2, fWhh, fW1, fW2, out);
}
